// RNN_66683662237893
// MI455X (gfx1250) — matrix operand order repaired and checked by reading, not compiled, not run on hardware
//
#include <hip/hip_runtime.h>
#include <hip/hip_bf16.h>

// Problem constants (match reference)
#define SEQ    128
#define BATCH  64
#define HID    1024
#define VOCAB  10000
#define NL     2

typedef __attribute__((ext_vector_type(16))) __bf16 v16bf;
typedef __attribute__((ext_vector_type(8)))  float  v8f;

// ---------------------------------------------------------------------------
// WMMA fragment loaders (wave32, v_wmma_f32_16x16x32_bf16; contiguous k order, not the ISA table order)
//
// A matrix 16x32 bf16 (M x K), row-major source A[m][k], lda = K stride:
//   lane 0-15 : M = lane,    K 0..15
//   lane 16-31: M = lane-16, K 16..31
//   VGPR v (pair 2*v,2*v+1): K = 2*v + 16*(lane>=16)
__device__ __forceinline__ v16bf load_a_frag(const __bf16* __restrict__ A,
                                             int lda, int m_base, int k_base,
                                             int lane) {
  const int m  = m_base + (lane & 15);
  const int kh = (lane >> 4) << 4;                // 0 or 16
  const __bf16* row = A + (size_t)m * lda + (k_base + kh);
  v16bf a;
#pragma unroll
  for (int v = 0; v < 8; ++v) {
    const int k = 2 * v;
    a[2 * v]     = row[k];
    a[2 * v + 1] = row[k + 1];
  }
  return a;
}

// B matrix 32x16 bf16 (K x N). Weights are PRE-TRANSPOSED (N x K row-major,
// Bt[n][k]) so each lane reads 16 contiguous bf16 (32B, aligned):
//   lane 0-15 : N = lane,    halves = K 0..15
//   lane 16-31: N = lane-16, halves = K 16..31
__device__ __forceinline__ v16bf load_b_frag(const __bf16* __restrict__ Bt,
                                             int ldb, int n_base, int k_base,
                                             int lane) {
  const int n  = n_base + (lane & 15);
  const int ks = k_base + ((lane >> 4) << 4);     // +0 or +16
  return *(const v16bf*)(Bt + (size_t)n * ldb + ks);
}

// ---------------------------------------------------------------------------
// Recurrent layer: Hnew = tanh(X*Wi + Hprev*Wh + bias)
// X, Hprev: (BATCH x HID) bf16 row-major.  Wti/Wth: (HID x HID) bf16, stored
// transposed [n][k].  256 output tiles (4 Mtiles x 64 Ntiles), 1 wave/tile,
// 8 waves/block -> grid = 32, exact (no divergence around WMMA). This kernel
// is launch-latency bound (sequential over 128 steps), so max parallelism
// beats register blocking here.
__global__ void __launch_bounds__(256) rnn_layer_kernel(
    const __bf16* __restrict__ X, const __bf16* __restrict__ Hprev,
    const __bf16* __restrict__ Wti, const __bf16* __restrict__ Wth,
    const float* __restrict__ bias,
    __bf16* __restrict__ Hnew, float* __restrict__ Hf32) {
  const int lane = threadIdx.x & 31;
  const int tile = blockIdx.x * 8 + (threadIdx.x >> 5);   // 0..255
  const int m0 = (tile >> 6) * 16;                        // BATCH/16 = 4 tiles
  const int n0 = (tile & 63) * 16;                        // HID/16  = 64 tiles

  v8f acc = {};
#pragma unroll 4
  for (int k = 0; k < HID; k += 32) {
    v16bf a0 = load_a_frag(X,     HID, m0, k, lane);
    v16bf b0 = load_b_frag(Wti,   HID, n0, k, lane);
    acc = __builtin_amdgcn_wmma_f32_16x16x32_bf16(false, a0, false, b0,
                                                  (short)0, acc, false, false);
    v16bf a1 = load_a_frag(Hprev, HID, m0, k, lane);
    v16bf b1 = load_b_frag(Wth,   HID, n0, k, lane);
    acc = __builtin_amdgcn_wmma_f32_16x16x32_bf16(false, a1, false, b1,
                                                  (short)0, acc, false, false);
  }

  // D layout: VGPR i -> M = m0 + 8*(lane>=16) + i, N = n0 + (lane&15)
  const int n    = n0 + (lane & 15);
  const int mrow = m0 + ((lane >> 4) << 3);
  const float b  = bias[n];
#pragma unroll
  for (int i = 0; i < 8; ++i) {
    float h = tanhf(acc[i] + b);
    Hnew[(size_t)(mrow + i) * HID + n] = (__bf16)h;
    Hf32[(size_t)(mrow + i) * HID + n] = h;      // survives to last step
  }
}

// ---------------------------------------------------------------------------
// Batched classifier: logits = H1all (8192 x 1024) * fcW (1024 x 10000) + fcb
// fcWt stored transposed (VOCAB x HID).
// Register-blocked 2M x 5N tiles per wave (32 x 80 output): per K-step,
// 2 A-frags + 5 B-frags (7 KB) feed 10 wmma (160 KFLOP) -> ~23 FLOP/B from
// L2 instead of 8.  Tile grid: (512/2) x (625/5) = 256 x 125 = 32000 waves,
// 8 waves/block -> grid = 4000, exact.
__global__ void __launch_bounds__(256) fc_kernel(
    const __bf16* __restrict__ H1, const __bf16* __restrict__ Wt,
    const float* __restrict__ bias, float* __restrict__ out) {
  const int lane = threadIdx.x & 31;
  const int tile = blockIdx.x * 8 + (threadIdx.x >> 5);
  const int NTB = (VOCAB / 16) / 5;                // 125 blocked N tiles
  const int m0 = (tile / NTB) * 32;
  const int n0 = (tile % NTB) * 80;

  v8f acc[2][5] = {};
#pragma unroll 2
  for (int k = 0; k < HID; k += 32) {
    v16bf a0 = load_a_frag(H1, HID, m0,      k, lane);
    v16bf a1 = load_a_frag(H1, HID, m0 + 16, k, lane);
#pragma unroll
    for (int j = 0; j < 5; ++j) {
      v16bf b = load_b_frag(Wt, HID, n0 + 16 * j, k, lane);
      acc[0][j] = __builtin_amdgcn_wmma_f32_16x16x32_bf16(
          false, a0, false, b, (short)0, acc[0][j], false, false);
      acc[1][j] = __builtin_amdgcn_wmma_f32_16x16x32_bf16(
          false, a1, false, b, (short)0, acc[1][j], false, false);
    }
  }

  // Epilogue: bias add, non-temporal store (328 MB written once, never
  // re-read -> keep it out of L2 so the hot weights stay resident).
  const int nlo  = lane & 15;
  const int mrow = (lane >> 4) << 3;
#pragma unroll
  for (int mi = 0; mi < 2; ++mi) {
#pragma unroll
    for (int j = 0; j < 5; ++j) {
      const int n  = n0 + 16 * j + nlo;
      const float bb = bias[n];
      float* col = out + (size_t)(m0 + 16 * mi + mrow) * VOCAB + n;
#pragma unroll
      for (int i = 0; i < 8; ++i)
        __builtin_nontemporal_store(acc[mi][j][i] + bb, col + (size_t)i * VOCAB);
    }
  }
}

// ---------------------------------------------------------------------------
// Prep / epilogue kernels
__global__ void transpose_cvt_kernel(const float* __restrict__ W,
                                     __bf16* __restrict__ Wt,
                                     int K, int N) {          // Wt[n][k] = W[k][n]
  long i = (long)blockIdx.x * blockDim.x + threadIdx.x;
  long total = (long)K * N;
  if (i >= total) return;
  int n = (int)(i / K), k = (int)(i % K);
  Wt[i] = (__bf16)W[(size_t)k * N + n];
}

__global__ void cvt_kernel(const float* __restrict__ in,
                           __bf16* __restrict__ out, long n) {
  long i = (long)blockIdx.x * blockDim.x + threadIdx.x;
  if (i < n) out[i] = (__bf16)in[i];
}

__global__ void embed_kernel(const int* __restrict__ tok,
                             const float* __restrict__ E,
                             __bf16* __restrict__ out) {       // (S*B, HID)
  long i = (long)blockIdx.x * blockDim.x + threadIdx.x;        // exact grid
  int  h  = (int)(i & (HID - 1));
  long sb = i >> 10;
  out[i] = (__bf16)E[(size_t)tok[sb] * HID + h];
}

__global__ void copy_f32_kernel(const float* __restrict__ in,
                                float* __restrict__ out, long n) {
  long i = (long)blockIdx.x * blockDim.x + threadIdx.x;
  if (i < n) out[i] = in[i];
}

// ---------------------------------------------------------------------------
extern "C" void kernel_launch(void* const* d_in, const int* in_sizes, int n_in,
                              void* d_out, int out_size, void* d_ws, size_t ws_size,
                              hipStream_t stream) {
  const int*   inputs = (const int*)  d_in[0];   // (S,B)
  const float* hidden = (const float*)d_in[1];   // (L,B,H)
  const float* embW   = (const float*)d_in[2];   // (V,H)
  const float* Wi     = (const float*)d_in[3];   // (L,H,H)
  const float* Wh     = (const float*)d_in[4];   // (L,H,H)
  const float* bh     = (const float*)d_in[5];   // (L,1,H)
  const float* fcW    = (const float*)d_in[6];   // (H,V)
  const float* fcb    = (const float*)d_in[7];   // (V)

  const size_t BH = (size_t)BATCH * HID;

  // Workspace layout (~80 MB, all offsets 256B-aligned)
  char* ws = (char*)d_ws;
  __bf16* embs  = (__bf16*)(ws);                                 // 16 MB (S*B*H)
  __bf16* h0all = (__bf16*)(ws + 16777216);                      // 16 MB
  __bf16* h1all = (__bf16*)(ws + 33554432);                      // 16 MB
  __bf16* Wti   = (__bf16*)(ws + 50331648);                      // 4 MB  (2 x HxH)
  __bf16* Wth   = (__bf16*)(ws + 54525952);                      // 4 MB
  __bf16* fcWt  = (__bf16*)(ws + 58720256);                      // 20 MB (V x H)
  __bf16* hinit = (__bf16*)(ws + 79200256);                      // 256 KB (L*B*H)
  float*  hlast = (float*) (ws + 79462400);                      // 512 KB (L*B*H f32)

  float* logits     = (float*)d_out;                             // (S*B, V)
  float* hidden_out = logits + (size_t)SEQ * BATCH * VOCAB;      // (L,B,H)

  // --- weight prep: convert + transpose to bf16 (reused 128x out of L2) ---
  for (int l = 0; l < NL; ++l) {
    transpose_cvt_kernel<<<(HID * HID + 255) / 256, 256, 0, stream>>>(
        Wi + (size_t)l * HID * HID, Wti + (size_t)l * HID * HID, HID, HID);
    transpose_cvt_kernel<<<(HID * HID + 255) / 256, 256, 0, stream>>>(
        Wh + (size_t)l * HID * HID, Wth + (size_t)l * HID * HID, HID, HID);
  }
  transpose_cvt_kernel<<<(HID * VOCAB + 255) / 256, 256, 0, stream>>>(
      fcW, fcWt, HID, VOCAB);
  cvt_kernel<<<(int)((NL * BH + 255) / 256), 256, 0, stream>>>(
      hidden, hinit, (long)NL * BH);
  embed_kernel<<<(int)(((size_t)SEQ * BH) / 256), 256, 0, stream>>>(
      inputs, embW, embs);

  // --- sequential recurrence: 2 WMMA kernels per timestep -----------------
  for (int t = 0; t < SEQ; ++t) {
    const __bf16* h0p = (t == 0) ? hinit      : h0all + (size_t)(t - 1) * BH;
    const __bf16* h1p = (t == 0) ? hinit + BH : h1all + (size_t)(t - 1) * BH;
    rnn_layer_kernel<<<32, 256, 0, stream>>>(
        embs + (size_t)t * BH, h0p, Wti, Wth, bh,
        h0all + (size_t)t * BH, hlast);
    rnn_layer_kernel<<<32, 256, 0, stream>>>(
        h0all + (size_t)t * BH, h1p,
        Wti + (size_t)HID * HID, Wth + (size_t)HID * HID, bh + HID,
        h1all + (size_t)t * BH, hlast + BH);
  }

  // --- hoisted classifier GEMM over all timesteps (parallel, WMMA) --------
  fc_kernel<<<4000, 256, 0, stream>>>(h1all, fcWt, fcb, logits);

  // --- final hidden state (f32 copies from the last step) -----------------
  copy_f32_kernel<<<(int)((NL * BH + 255) / 256), 256, 0, stream>>>(
      hlast, hidden_out, (long)NL * BH);
}
